// GraphSAGE_16870631539213
// MI455X (gfx1250) — matrix operand order repaired and checked by reading, not compiled, not run on hardware
//
#include <hip/hip_runtime.h>
#include <hip/hip_bf16.h>
#include <math.h>

// ---------------------------------------------------------------------------
// GraphSAGE (3x SAGEConv+BN+ReLU) + attention pooling + MLP head for gfx1250.
// GEMMs run on v_wmma_f32_16x16x32_bf16 (bf16 in, f32 accumulate).
// Weights are pre-transposed to [col][K] so B fragments come straight from
// global memory (L2-resident); A tiles are staged in LDS, via async
// global->LDS loads (ASYNCcnt) when the toolchain exposes the builtins.
// ---------------------------------------------------------------------------

typedef __attribute__((ext_vector_type(16))) __bf16 v16bf;
typedef __attribute__((ext_vector_type(8)))  __bf16 bf16x8;
typedef __attribute__((ext_vector_type(8)))  float  v8f;

#if __has_builtin(__builtin_amdgcn_global_load_async_to_lds_b128) && \
    __has_builtin(__builtin_amdgcn_s_wait_asynccnt)
#define USE_ASYNC_LDS 1
typedef int i32x4 __attribute__((ext_vector_type(4)));
typedef __attribute__((address_space(1))) i32x4* gvec_t;   // global int4*
typedef __attribute__((address_space(3))) i32x4* lvec_t;   // LDS int4*
#else
#define USE_ASYNC_LDS 0
#endif

#define NNODES 100000
#define NEDGES 400000
#define INDIM  771
#define HID    256
#define NGR    64
#define K0P    800   // 771 padded to multiple of 32

// ------------------------------- utility -----------------------------------

__global__ void k_zero_f32(float* __restrict__ p, long n) {
  long i = (long)blockIdx.x * blockDim.x + threadIdx.x;
  if (i < n) p[i] = 0.0f;
}

__global__ void k_fill_f32(float* __restrict__ p, float v, long n) {
  long i = (long)blockIdx.x * blockDim.x + threadIdx.x;
  if (i < n) p[i] = v;
}

// ------------------------------ graph aggregation ---------------------------

__global__ void k_degree(const int* __restrict__ dst, float* __restrict__ deg, int E) {
  int e = blockIdx.x * blockDim.x + threadIdx.x;
  if (e < E) atomicAdd(&deg[dst[e]], 1.0f);
}

// one block per edge; threads stride features -> coalesced
__global__ __launch_bounds__(256) void k_scatter(
    const float* __restrict__ h, const int* __restrict__ src,
    const int* __restrict__ dst, float* __restrict__ msum, int D) {
  int e = blockIdx.x;
  long sofs = (long)src[e] * D;
  long dofs = (long)dst[e] * D;
  for (int i = threadIdx.x; i < D; i += 256)
    atomicAdd(&msum[dofs + i], h[sofs + i]);
}

// agg = msum / max(deg,1), converted to bf16 into padded [N, KP] buffer
__global__ void k_norm_conv(const float* __restrict__ msum, const float* __restrict__ deg,
                            __bf16* __restrict__ aggB, int N, int D, int KP) {
  long idx = (long)blockIdx.x * blockDim.x + threadIdx.x;
  if (idx >= (long)N * KP) return;
  int n = (int)(idx / KP);
  int p = (int)(idx % KP);
  float v = 0.0f;
  if (p < D) {
    float dg = deg[n];
    v = msum[(long)n * D + p] / fmaxf(dg, 1.0f);
  }
  aggB[idx] = (__bf16)v;
}

// dense f32 [N,D] -> bf16 padded [N,KP]
__global__ void k_conv_h(const float* __restrict__ h, __bf16* __restrict__ hB,
                         int N, int D, int KP) {
  long idx = (long)blockIdx.x * blockDim.x + threadIdx.x;
  if (idx >= (long)N * KP) return;
  int n = (int)(idx / KP);
  int p = (int)(idx % KP);
  hB[idx] = (__bf16)((p < D) ? h[(long)n * D + p] : 0.0f);
}

// weights f32 [K,256] row-major -> bf16 TRANSPOSED [256][KP], zero pad
__global__ void k_conv_wt(const float* __restrict__ W, __bf16* __restrict__ WT,
                          int K, int KP) {
  long idx = (long)blockIdx.x * blockDim.x + threadIdx.x;
  if (idx >= (long)KP * HID) return;
  int c = (int)(idx / KP);   // output column (0..255)
  int k = (int)(idx % KP);   // K index (coalesced writes)
  WT[idx] = (__bf16)((k < K) ? W[(long)k * HID + c] : 0.0f);
}

// ------------------------------ dual WMMA GEMM ------------------------------
// out[N,256] = A1 @ W1 + A2 @ W2 + bias
//   A1,A2: [N,KP] bf16 (LDS-staged);  W1t,W2t: [256][KP] bf16 (global, L2-hot)
// Block tile 128x32: 8 waves, each wave owns 16 rows x 32 cols (2 accumulators,
// A fragments reused across both column sub-tiles) -> 4 WMMA per wave per k-step.
__global__ __launch_bounds__(256) void k_gemm_dual(
    const __bf16* __restrict__ A1, const __bf16* __restrict__ A2,
    const __bf16* __restrict__ W1t, const __bf16* __restrict__ W2t,
    const float* __restrict__ bias, float* __restrict__ out, int N, int KP) {
  __shared__ __bf16 sA1[128][32];
  __shared__ __bf16 sA2[128][32];

  const int tid  = threadIdx.x;
  const int lane = tid & 31;
  const int wave = tid >> 5;                 // 0..7 -> row sub-tile
  const long rowBase = (long)blockIdx.x * 128;
  const int  colBase = blockIdx.y * 32;

  const int m    = lane & 15;
  const int half = lane >> 4;

  // per-lane weight row pointers (transposed layout: [col][K])
  const __bf16* w1c0 = W1t + (long)(colBase + m) * KP;
  const __bf16* w1c1 = W1t + (long)(colBase + 16 + m) * KP;
  const __bf16* w2c0 = W2t + (long)(colBase + m) * KP;
  const __bf16* w2c1 = W2t + (long)(colBase + 16 + m) * KP;

  // A-tile cooperative load mapping: 2 threads per row, 16 bf16 each
  const int  ar = tid >> 1;
  const int  aq = (tid & 1) * 16;
  const long gr = rowBase + ar;

  v8f c0 = {};
  v8f c1 = {};

  for (int k0 = 0; k0 < KP; k0 += 32) {
    __syncthreads();
    if (gr < N) {
      const __bf16* ap1 = &A1[gr * KP + k0 + aq];
      const __bf16* ap2 = &A2[gr * KP + k0 + aq];
#if USE_ASYNC_LDS
      // direct global->LDS DMA, tracked by ASYNCcnt (no VGPR transit)
      __builtin_amdgcn_global_load_async_to_lds_b128(
          (gvec_t)ap1,       (lvec_t)&sA1[ar][aq],     0, 0);
      __builtin_amdgcn_global_load_async_to_lds_b128(
          (gvec_t)(ap1 + 8), (lvec_t)&sA1[ar][aq + 8], 0, 0);
      __builtin_amdgcn_global_load_async_to_lds_b128(
          (gvec_t)ap2,       (lvec_t)&sA2[ar][aq],     0, 0);
      __builtin_amdgcn_global_load_async_to_lds_b128(
          (gvec_t)(ap2 + 8), (lvec_t)&sA2[ar][aq + 8], 0, 0);
#else
      *(bf16x8*)&sA1[ar][aq]     = *(const bf16x8*)(ap1);
      *(bf16x8*)&sA1[ar][aq + 8] = *(const bf16x8*)(ap1 + 8);
      *(bf16x8*)&sA2[ar][aq]     = *(const bf16x8*)(ap2);
      *(bf16x8*)&sA2[ar][aq + 8] = *(const bf16x8*)(ap2 + 8);
#endif
      if (k0 + 32 < KP) {       // prefetch next k-tile (global_prefetch_b8)
        __builtin_prefetch(ap1 + 32, 0, 3);
        __builtin_prefetch(ap2 + 32, 0, 3);
      }
    } else {
      bf16x8 z = {};
      *(bf16x8*)&sA1[ar][aq]     = z;
      *(bf16x8*)&sA1[ar][aq + 8] = z;
      *(bf16x8*)&sA2[ar][aq]     = z;
      *(bf16x8*)&sA2[ar][aq + 8] = z;
    }
#if USE_ASYNC_LDS
    __builtin_amdgcn_s_wait_asynccnt(0);
#endif
    __syncthreads();

    // --- A fragments (contiguous 16 K per lane, matching B) ---
    const int arow = wave * 16 + m;
    bf16x8 a1lo = *(const bf16x8*)&sA1[arow][half * 16];
    bf16x8 a1hi = *(const bf16x8*)&sA1[arow][half * 16 + 8];
    bf16x8 a2lo = *(const bf16x8*)&sA2[arow][half * 16];
    bf16x8 a2hi = *(const bf16x8*)&sA2[arow][half * 16 + 8];
    v16bf a1 = __builtin_shufflevector(a1lo, a1hi, 0,1,2,3,4,5,6,7,8,9,10,11,12,13,14,15);
    v16bf a2 = __builtin_shufflevector(a2lo, a2hi, 0,1,2,3,4,5,6,7,8,9,10,11,12,13,14,15);

    // --- B fragments straight from global (contiguous 16 K per lane) ---
    const int kb = k0 + half * 16;
    bf16x8 p0, p1;
    p0 = *(const bf16x8*)&w1c0[kb]; p1 = *(const bf16x8*)&w1c0[kb + 8];
    v16bf b1c0 = __builtin_shufflevector(p0, p1, 0,1,2,3,4,5,6,7,8,9,10,11,12,13,14,15);
    p0 = *(const bf16x8*)&w1c1[kb]; p1 = *(const bf16x8*)&w1c1[kb + 8];
    v16bf b1c1 = __builtin_shufflevector(p0, p1, 0,1,2,3,4,5,6,7,8,9,10,11,12,13,14,15);
    p0 = *(const bf16x8*)&w2c0[kb]; p1 = *(const bf16x8*)&w2c0[kb + 8];
    v16bf b2c0 = __builtin_shufflevector(p0, p1, 0,1,2,3,4,5,6,7,8,9,10,11,12,13,14,15);
    p0 = *(const bf16x8*)&w2c1[kb]; p1 = *(const bf16x8*)&w2c1[kb + 8];
    v16bf b2c1 = __builtin_shufflevector(p0, p1, 0,1,2,3,4,5,6,7,8,9,10,11,12,13,14,15);

    c0 = __builtin_amdgcn_wmma_f32_16x16x32_bf16(false, a1, false, b1c0, (short)0, c0, false, false);
    c0 = __builtin_amdgcn_wmma_f32_16x16x32_bf16(false, a2, false, b2c0, (short)0, c0, false, false);
    c1 = __builtin_amdgcn_wmma_f32_16x16x32_bf16(false, a1, false, b1c1, (short)0, c1, false, false);
    c1 = __builtin_amdgcn_wmma_f32_16x16x32_bf16(false, a2, false, b2c1, (short)0, c1, false, false);
  }

  // --- epilogue: C layout VGPR i -> row = i + 8*half, col = lane&15 ---
  const int col0 = colBase + m;
  const int col1 = colBase + 16 + m;
  const float bia0 = bias[col0];
  const float bia1 = bias[col1];
  #pragma unroll
  for (int i = 0; i < 8; ++i) {
    long row = rowBase + wave * 16 + 8 * half + i;
    if (row < N) {
      out[row * HID + col0] = c0[i] + bia0;
      out[row * HID + col1] = c1[i] + bia1;
    }
  }
}

// ------------------------------ batch-norm ----------------------------------

__global__ __launch_bounds__(256) void k_bn_stats(const float* __restrict__ out,
                                                  float* __restrict__ sum,
                                                  float* __restrict__ sumsq, int N) {
  int j = threadIdx.x;     // column
  float s = 0.0f, q = 0.0f;
  for (int r = blockIdx.x; r < N; r += gridDim.x) {
    float v = out[(long)r * HID + j];
    s += v; q += v * v;
  }
  atomicAdd(&sum[j], s);
  atomicAdd(&sumsq[j], q);
}

__global__ void k_bn_apply(const float* __restrict__ out, const float* __restrict__ sum,
                           const float* __restrict__ sumsq, const float* __restrict__ g,
                           const float* __restrict__ b, float* __restrict__ hF,
                           __bf16* __restrict__ hB, int N) {
  long idx = (long)blockIdx.x * blockDim.x + threadIdx.x;
  if (idx >= (long)N * HID) return;
  int d = (int)(idx & (HID - 1));
  const float invN = 1.0f / (float)N;
  float mean = sum[d] * invN;
  float var  = sumsq[d] * invN - mean * mean;
  float v = (out[idx] - mean) * rsqrtf(var + 1e-5f) * g[d] + b[d];
  v = v > 0.0f ? v : 0.0f;
  hF[idx] = v;
  hB[idx] = (__bf16)v;   // next layer GEMM input (stride 256)
}

// --------------------------- attention pooling ------------------------------

__global__ __launch_bounds__(128) void k_score(const float* __restrict__ h,
                                               const float* __restrict__ Wa1,
                                               const float* __restrict__ ba1,
                                               const float* __restrict__ Wa2,
                                               const float* __restrict__ ba2,
                                               float* __restrict__ s, int N) {
  __shared__ float sh[HID];
  __shared__ float red[128];
  int n = blockIdx.x;
  int t = threadIdx.x;
  sh[t]       = h[(long)n * HID + t];
  sh[t + 128] = h[(long)n * HID + t + 128];
  __syncthreads();
  float acc = 0.0f;
  for (int k = 0; k < HID; ++k) acc += sh[k] * Wa1[k * 128 + t];
  red[t] = tanhf(acc + ba1[t]) * Wa2[t];
  __syncthreads();
  for (int off = 64; off > 0; off >>= 1) {
    if (t < off) red[t] += red[t + off];
    __syncthreads();
  }
  if (t == 0) s[n] = red[0] + ba2[0];
}

__device__ inline void atomicMaxF(float* addr, float val) {
  unsigned int* ua = (unsigned int*)addr;
  unsigned int old = *ua;
  while (__uint_as_float(old) < val) {
    unsigned int assumed = old;
    old = atomicCAS(ua, assumed, __float_as_uint(val));
    if (old == assumed) break;
  }
}

__global__ void k_seg_max(const float* __restrict__ s, const int* __restrict__ batch,
                          float* __restrict__ gmax, int N) {
  int n = blockIdx.x * blockDim.x + threadIdx.x;
  if (n < N) atomicMaxF(&gmax[batch[n]], s[n]);
}

__global__ void k_exp_sum(float* __restrict__ s, const int* __restrict__ batch,
                          const float* __restrict__ gmax, float* __restrict__ gden, int N) {
  int n = blockIdx.x * blockDim.x + threadIdx.x;
  if (n >= N) return;
  int b = batch[n];
  float e = expf(s[n] - gmax[b]);
  s[n] = e;                       // overwrite with exp value
  atomicAdd(&gden[b], e);
}

__global__ void k_pool(const float* __restrict__ h, const float* __restrict__ sE,
                       const float* __restrict__ gden, const int* __restrict__ batch,
                       float* __restrict__ pooled, int N) {
  long idx = (long)blockIdx.x * blockDim.x + threadIdx.x;
  if (idx >= (long)N * HID) return;
  int n = (int)(idx >> 8);
  int d = (int)(idx & (HID - 1));
  int b = batch[n];
  float w = sE[n] / gden[b];
  atomicAdd(&pooled[(long)b * HID + d], h[idx] * w);
}

__global__ __launch_bounds__(128) void k_classifier(const float* __restrict__ pooled,
                                                    const float* __restrict__ Wc1,
                                                    const float* __restrict__ bc1,
                                                    const float* __restrict__ Wc2,
                                                    const float* __restrict__ bc2,
                                                    float* __restrict__ out) {
  __shared__ float sp[HID];
  __shared__ float red[128];
  int g = blockIdx.x;
  int t = threadIdx.x;
  sp[t]       = pooled[(long)g * HID + t];
  sp[t + 128] = pooled[(long)g * HID + t + 128];
  __syncthreads();
  float acc = 0.0f;
  for (int k = 0; k < HID; ++k) acc += sp[k] * Wc1[k * 128 + t];
  acc += bc1[t];
  acc = acc > 0.0f ? acc : 0.0f;
  red[t] = acc * Wc2[t];
  __syncthreads();
  for (int off = 64; off > 0; off >>= 1) {
    if (t < off) red[t] += red[t + off];
    __syncthreads();
  }
  if (t == 0) out[g] = red[0] + bc2[0];
}

// ------------------------------- host side ----------------------------------

static inline int cdiv(long a, long b) { return (int)((a + b - 1) / b); }

extern "C" void kernel_launch(void* const* d_in, const int* in_sizes, int n_in,
                              void* d_out, int out_size, void* d_ws, size_t ws_size,
                              hipStream_t stream) {
  (void)in_sizes; (void)n_in; (void)out_size; (void)ws_size;
  const int N = NNODES, E = NEDGES;

  const float* x     = (const float*)d_in[0];
  const int*   ei    = (const int*)d_in[1];
  const int*   batch = (const int*)d_in[2];
  const float* Wl[3] = {(const float*)d_in[3],  (const float*)d_in[8],  (const float*)d_in[13]};
  const float* bl[3] = {(const float*)d_in[4],  (const float*)d_in[9],  (const float*)d_in[14]};
  const float* Wr[3] = {(const float*)d_in[5],  (const float*)d_in[10], (const float*)d_in[15]};
  const float* gm[3] = {(const float*)d_in[6],  (const float*)d_in[11], (const float*)d_in[16]};
  const float* bt[3] = {(const float*)d_in[7],  (const float*)d_in[12], (const float*)d_in[17]};
  const float* Wa1 = (const float*)d_in[18];
  const float* ba1 = (const float*)d_in[19];
  const float* Wa2 = (const float*)d_in[20];
  const float* ba2 = (const float*)d_in[21];
  const float* Wc1 = (const float*)d_in[22];
  const float* bc1 = (const float*)d_in[23];
  const float* Wc2 = (const float*)d_in[24];
  const float* bc2 = (const float*)d_in[25];

  const int* src = ei;
  const int* dst = ei + E;

  // workspace carve (256B aligned slices)
  char*  base = (char*)d_ws;
  size_t off  = 0;
  auto alloc = [&](size_t bytes) -> void* {
    void* p = base + off;
    off = (off + bytes + 255) & ~(size_t)255;
    return p;
  };
  float*  deg    = (float*)alloc((size_t)N * 4);
  float*  msum   = (float*)alloc((size_t)N * K0P * 4);
  __bf16* aggB   = (__bf16*)alloc((size_t)N * K0P * 2);
  __bf16* hB     = (__bf16*)alloc((size_t)N * K0P * 2);
  float*  hF     = (float*)alloc((size_t)N * HID * 4);
  float*  outF   = (float*)alloc((size_t)N * HID * 4);
  __bf16* WlT    = (__bf16*)alloc((size_t)K0P * HID * 2);   // [256][KP]
  __bf16* WrT    = (__bf16*)alloc((size_t)K0P * HID * 2);   // [256][KP]
  float*  stats  = (float*)alloc(512 * 4);                  // sum[256] | sumsq[256]
  float*  sArr   = (float*)alloc((size_t)N * 4);
  float*  gmax   = (float*)alloc(NGR * 4);
  float*  gden   = (float*)alloc(NGR * 4);
  float*  pooled = (float*)alloc((size_t)NGR * HID * 4);

  // degree (once)
  k_zero_f32<<<cdiv(N, 256), 256, 0, stream>>>(deg, N);
  k_degree<<<cdiv(E, 256), 256, 0, stream>>>(dst, deg, E);

  const int Darr[3]  = {INDIM, HID, HID};
  const int KParr[3] = {K0P, HID, HID};

  for (int i = 0; i < 3; ++i) {
    const int D = Darr[i], KP = KParr[i];
    const float* hIn = (i == 0) ? x : hF;

    k_zero_f32<<<cdiv((long)N * D, 256), 256, 0, stream>>>(msum, (long)N * D);
    k_scatter<<<E, 256, 0, stream>>>(hIn, src, dst, msum, D);
    k_norm_conv<<<cdiv((long)N * KP, 256), 256, 0, stream>>>(msum, deg, aggB, N, D, KP);
    k_conv_wt<<<cdiv((long)KP * HID, 256), 256, 0, stream>>>(Wl[i], WlT, D, KP);
    k_conv_wt<<<cdiv((long)KP * HID, 256), 256, 0, stream>>>(Wr[i], WrT, D, KP);
    if (i == 0)
      k_conv_h<<<cdiv((long)N * KP, 256), 256, 0, stream>>>(x, hB, N, D, KP);

    dim3 gg(cdiv(N, 128), HID / 32);
    k_gemm_dual<<<gg, 256, 0, stream>>>(aggB, hB, WlT, WrT, bl[i], outF, N, KP);

    k_zero_f32<<<cdiv(512, 256), 256, 0, stream>>>(stats, 512);
    k_bn_stats<<<512, 256, 0, stream>>>(outF, stats, stats + 256, N);
    k_bn_apply<<<cdiv((long)N * HID, 256), 256, 0, stream>>>(
        outF, stats, stats + 256, gm[i], bt[i], hF, hB, N);
  }

  // attention pooling
  k_score<<<N, 128, 0, stream>>>(hF, Wa1, ba1, Wa2, ba2, sArr, N);
  k_fill_f32<<<1, 64, 0, stream>>>(gmax, -3.0e38f, NGR);
  k_zero_f32<<<1, 64, 0, stream>>>(gden, NGR);
  k_zero_f32<<<cdiv((long)NGR * HID, 256), 256, 0, stream>>>(pooled, (long)NGR * HID);
  k_seg_max<<<cdiv(N, 256), 256, 0, stream>>>(sArr, batch, gmax, N);
  k_exp_sum<<<cdiv(N, 256), 256, 0, stream>>>(sArr, batch, gmax, gden, N);
  k_pool<<<cdiv((long)N * HID, 256), 256, 0, stream>>>(hF, sArr, gden, batch, pooled, N);

  // classifier head -> d_out[64]
  k_classifier<<<NGR, 128, 0, stream>>>(pooled, Wc1, bc1, Wc2, bc2, (float*)d_out);
}
